// GRULayer_78460462563561
// MI455X (gfx1250) — compile-verified, matrix operand order checked by reading, not run on hardware
//
#include <hip/hip_runtime.h>
#include <hip/hip_bf16.h>
#include <stdint.h>

#define SEQ 512
#define BATCH 64
#define ISZ 1024
#define HSZ 1024
#define MTOT (SEQ * BATCH)

typedef __attribute__((ext_vector_type(16))) __bf16 v16bf;
typedef __attribute__((ext_vector_type(8)))  float  v8f;
typedef __attribute__((ext_vector_type(4)))  unsigned int u32x4;
typedef __attribute__((ext_vector_type(8)))  unsigned int u32x8;

union Frag16 { unsigned int u[8]; v16bf v; };

__device__ __forceinline__ unsigned short f32_to_bf16_rne(float f) {
    unsigned int u = __float_as_uint(f);
    u += 0x7FFFu + ((u >> 16) & 1u);   // round-to-nearest-even
    return (unsigned short)(u >> 16);
}

__device__ __forceinline__ unsigned int pack_bf16x2(float lo, float hi) {
    return (unsigned int)f32_to_bf16_rne(lo) |
           ((unsigned int)f32_to_bf16_rne(hi) << 16);
}

__device__ __forceinline__ float sigmoidf_fast(float x) {
    return 1.0f / (1.0f + __expf(-x));
}

// ---------------------------------------------------------------------------
// Tensor Data Mover: 2D tile load Global -> LDS (D# groups 0/1 only).
// Issued per-wave (EXEC ignored, TENSORcnt-tracked).
// data_size_code: 0=1B 1=2B 2=4B 3=8B.  Dims/strides in elements.
// ---------------------------------------------------------------------------
__device__ __forceinline__ void tdm_load_2d_to_lds(
    unsigned lds_byte_addr, const void* tile_gaddr,
    unsigned data_size_code,
    unsigned tensor_d0, unsigned tensor_d1,
    unsigned tile_d0, unsigned tile_d1,
    unsigned long long stride_d0)
{
    unsigned long long ga = (unsigned long long)tile_gaddr;
    u32x4 g0;
    g0[0] = 1u;                                            // count=1, load, user
    g0[1] = lds_byte_addr;                                 // lds_addr
    g0[2] = (unsigned)ga;                                  // global_addr[31:0]
    g0[3] = (unsigned)((ga >> 32) & 0x01FFFFFFu)           // global_addr[56:32]
          | (2u << 30);                                    // type = 2 ("image")
    u32x8 g1;
    g1[0] = (data_size_code & 3u) << 16;                   // data_size
    g1[1] = (tensor_d0 & 0xFFFFu) << 16;                   // tensor_dim0[15:0]
    g1[2] = ((tensor_d0 >> 16) & 0xFFFFu)                  // tensor_dim0[31:16]
          | ((tensor_d1 & 0xFFFFu) << 16);                 // tensor_dim1[15:0]
    g1[3] = ((tensor_d1 >> 16) & 0xFFFFu)                  // tensor_dim1[31:16]
          | ((tile_d0 & 0xFFFFu) << 16);                   // tile_dim0
    g1[4] = (tile_d1 & 0xFFFFu);                           // tile_dim1 (dim2=0)
    g1[5] = (unsigned)(stride_d0 & 0xFFFFFFFFull);         // dim0_stride[31:0]
    g1[6] = (unsigned)((stride_d0 >> 32) & 0xFFFFull);     // dim0_stride[47:32]
    g1[7] = 0u;                                            // dim1_stride (2D)
    asm volatile("tensor_load_to_lds %0, %1"
                 :: "s"(g0), "s"(g1)
                 : "memory");
}

__device__ __forceinline__ void tdm_wait0() {
    asm volatile("s_wait_tensorcnt 0x0" ::: "memory");
}

// u32 column within a 32-wide K chunk for VGPR index p, lane-half hi
// (documented 16-bit A/B WMMA fragment layout).
__device__ __forceinline__ int frag_c2(int p, int hi) {
    return (p & 3) + 4 * hi + ((p >= 4) ? 8 : 0);
}

// ---------------------------------------------------------------------------
// One-time conversion: input f32 -> bf16
// ---------------------------------------------------------------------------
__global__ void k_f32_to_bf16(const float* __restrict__ src,
                              unsigned short* __restrict__ dst, int n) {
    int i = blockIdx.x * blockDim.x + threadIdx.x;
    int stride = gridDim.x * blockDim.x;
    for (; i < n; i += stride) dst[i] = f32_to_bf16_rne(src[i]);
}

// ---------------------------------------------------------------------------
// One-time: split W{r,u,n} (H x (I+H)) into bf16 Wx (H x I) and Wh (H x H)
// ---------------------------------------------------------------------------
__global__ void k_split_weights(const float* __restrict__ Wr,
                                const float* __restrict__ Wu,
                                const float* __restrict__ Wn,
                                unsigned short* __restrict__ wx,   // [3][H][I]
                                unsigned short* __restrict__ wh) { // [3][H][H]
    const long long per = (long long)HSZ * (ISZ + HSZ);
    const long long total = 3LL * per;
    long long stride = (long long)gridDim.x * blockDim.x;
    for (long long i = (long long)blockIdx.x * blockDim.x + threadIdx.x;
         i < total; i += stride) {
        int g = (int)(i / per);
        long long r = i % per;
        int n = (int)(r / (ISZ + HSZ));
        int c = (int)(r % (ISZ + HSZ));
        const float* W = (g == 0) ? Wr : ((g == 1) ? Wu : Wn);
        unsigned short b = f32_to_bf16_rne(W[(size_t)n * (ISZ + HSZ) + c]);
        if (c < ISZ)
            wx[(size_t)g * HSZ * ISZ + (size_t)n * ISZ + c] = b;
        else
            wh[(size_t)g * HSZ * HSZ + (size_t)n * HSZ + (c - ISZ)] = b;
    }
}

// ---------------------------------------------------------------------------
// Pre-activation GEMM: X[g] = input @ Wx[g]^T + bias[g]
// Block tile 32(M) x 64(N), 4 waves, 2 accumulators/wave.
// Double-buffered TDM staging: DMA of chunk c+1 overlaps WMMA on chunk c.
// ---------------------------------------------------------------------------
__global__ __launch_bounds__(128) void k_gemm_xgates(
    const unsigned short* __restrict__ in_bf16,  // [MTOT][ISZ] bf16
    const unsigned short* __restrict__ wx,       // [3][HSZ][ISZ] bf16
    const float* __restrict__ br, const float* __restrict__ bu,
    const float* __restrict__ bn,
    float* __restrict__ X)                       // [3][MTOT][HSZ] f32
{
    __shared__ __align__(16) unsigned short ash[2][32 * 32];  // A ping-pong
    __shared__ __align__(16) unsigned short bsh[2][64 * 32];  // B ping-pong

    const int g  = blockIdx.z;
    const int m0 = blockIdx.y * 32;
    const int n0 = blockIdx.x * 64;
    const int tid  = threadIdx.x;
    const int lane = tid & 31;
    const int wv   = __builtin_amdgcn_readfirstlane(tid) >> 5;  // scalar wave id
    const int lo = lane & 15, hi = lane >> 4;
    const unsigned short* wg = wx + (size_t)g * HSZ * ISZ;

    v8f acc0 = {}, acc1 = {};

    const int NCH = ISZ / 32;
    // Prologue: stage chunk 0.
    if (wv == 0) {
        tdm_load_2d_to_lds((unsigned)(size_t)&ash[0][0],
                           in_bf16 + (size_t)m0 * ISZ,
                           /*2B*/1, ISZ, MTOT, 32, 32, ISZ);
    } else if (wv == 1) {
        tdm_load_2d_to_lds((unsigned)(size_t)&bsh[0][0],
                           wg + (size_t)n0 * ISZ,
                           /*2B*/1, ISZ, HSZ, 32, 64, ISZ);
    }

    for (int c = 0; c < NCH; c++) {
        const int buf = c & 1;
        if (wv < 2) tdm_wait0();        // chunk c landed (in-order TENSORcnt)
        __syncthreads();                // publish chunk c; prior reads done
        if (c + 1 < NCH) {              // overlap: stage chunk c+1
            const int k1 = (c + 1) * 32;
            if (wv == 0) {
                tdm_load_2d_to_lds((unsigned)(size_t)&ash[buf ^ 1][0],
                                   in_bf16 + (size_t)m0 * ISZ + k1,
                                   1, ISZ, MTOT, 32, 32, ISZ);
            } else if (wv == 1) {
                tdm_load_2d_to_lds((unsigned)(size_t)&bsh[buf ^ 1][0],
                                   wg + (size_t)n0 * ISZ + k1,
                                   1, ISZ, HSZ, 32, 64, ISZ);
            }
        }

        Frag16 a0, a1, b;
        const unsigned int* a32 = (const unsigned int*)ash[buf]; // [32][16] u32
        const unsigned int* b32 = (const unsigned int*)bsh[buf]; // [64][16] u32
        #pragma unroll
        for (int p = 0; p < 8; p++) {
            int c2 = frag_c2(p, hi);
            a0.u[p] = a32[lo * 16 + c2];
            a1.u[p] = a32[(lo + 16) * 16 + c2];
            b.u[p]  = b32[(wv * 16 + lo) * 16 + c2];
        }
        acc0 = __builtin_amdgcn_wmma_f32_16x16x32_bf16(
            false, a0.v, false, b.v, (short)0, acc0, false, false);
        acc1 = __builtin_amdgcn_wmma_f32_16x16x32_bf16(
            false, a1.v, false, b.v, (short)0, acc1, false, false);
        __syncthreads();                // reads done before buf is overwritten
    }

    const float* bias = (g == 0) ? br : ((g == 1) ? bu : bn);
    float* Xg = X + (size_t)g * MTOT * HSZ;
    const int n = n0 + wv * 16 + lo;
    const float bn_ = bias[n];
    #pragma unroll
    for (int r = 0; r < 8; r++) {
        int m = m0 + r + 8 * hi;
        Xg[(size_t)m * HSZ + n]        = acc0[r] + bn_;
        Xg[(size_t)(m + 16) * HSZ + n] = acc1[r] + bn_;
    }
}

// ---------------------------------------------------------------------------
// One recurrent step. Tile: 16 batch x 64 hidden, K-chunks of 64, double-
// buffered TDM staging (waves 0..2: one gate weight tile each; wave 3: f32 h
// tile, converted to bf16 at fragment build so the recurrence state stays f32).
// ---------------------------------------------------------------------------
__global__ __launch_bounds__(128) void k_gru_step(
    const float* __restrict__ h_in,        // [BATCH][HSZ] f32
    const unsigned short* __restrict__ wh, // [3][HSZ][HSZ] bf16
    const float* __restrict__ Xr, const float* __restrict__ Xu,
    const float* __restrict__ Xn,          // [BATCH][HSZ] (this step)
    float* __restrict__ out_s,             // [BATCH][HSZ]
    float* __restrict__ h_out)             // [BATCH][HSZ]
{
    __shared__ __align__(16) float          ashf[2][16 * 64];     // 2 x 4KB
    __shared__ __align__(16) unsigned short bsh[2][3][64 * 64];   // 2 x 24KB

    const int b0 = blockIdx.y * 16;
    const int n0 = blockIdx.x * 64;
    const int tid  = threadIdx.x;
    const int lane = tid & 31;
    const int wv   = __builtin_amdgcn_readfirstlane(tid) >> 5;  // scalar wave id
    const int lo = lane & 15, hi = lane >> 4;

    // Warm the pre-activation lines needed in the epilogue.
    __builtin_prefetch(Xr + (size_t)b0 * HSZ + n0, 0, 3);
    __builtin_prefetch(Xn + (size_t)b0 * HSZ + n0, 0, 3);

    v8f accR = {}, accU = {}, accN = {};

    const int NCH = HSZ / 64;
    // Prologue: stage chunk 0.
    if (wv < 3) {
        tdm_load_2d_to_lds((unsigned)(size_t)&bsh[0][wv][0],
                           wh + (size_t)wv * HSZ * HSZ + (size_t)n0 * HSZ,
                           /*2B*/1, HSZ, HSZ, 64, 64, HSZ);
    } else {
        tdm_load_2d_to_lds((unsigned)(size_t)&ashf[0][0],
                           h_in + (size_t)b0 * HSZ,
                           /*4B*/2, HSZ, BATCH, 64, 16, HSZ);
    }

    for (int c = 0; c < NCH; c++) {
        const int buf = c & 1;
        tdm_wait0();                    // every wave issued one TDM per chunk
        __syncthreads();
        if (c + 1 < NCH) {              // overlap: stage chunk c+1
            const int k1 = (c + 1) * 64;
            if (wv < 3) {
                tdm_load_2d_to_lds((unsigned)(size_t)&bsh[buf ^ 1][wv][0],
                                   wh + (size_t)wv * HSZ * HSZ
                                      + (size_t)n0 * HSZ + k1,
                                   1, HSZ, HSZ, 64, 64, HSZ);
            } else {
                tdm_load_2d_to_lds((unsigned)(size_t)&ashf[buf ^ 1][0],
                                   h_in + (size_t)b0 * HSZ + k1,
                                   2, HSZ, BATCH, 64, 16, HSZ);
            }
        }

        const float* af = ashf[buf];
        const unsigned int* r32 = (const unsigned int*)bsh[buf][0];
        const unsigned int* u32p = (const unsigned int*)bsh[buf][1];
        const unsigned int* n32 = (const unsigned int*)bsh[buf][2];
        #pragma unroll
        for (int ks = 0; ks < 64; ks += 32) {
            // A fragment: f32 pairs from LDS -> packed bf16.
            Frag16 a;
            #pragma unroll
            for (int p = 0; p < 8; p++) {
                int kk = ks + 2 * (p & 3) + 8 * hi + ((p >= 4) ? 16 : 0);
                float2 hv = *(const float2*)&af[lo * 64 + kk];
                a.u[p] = pack_bf16x2(hv.x, hv.y);
            }
            Frag16 bR, bU, bN;
            const int brow = (wv * 16 + lo) * 32 + (ks >> 1); // row = 32 u32
            #pragma unroll
            for (int p = 0; p < 8; p++) {
                int c2 = frag_c2(p, hi);
                bR.u[p] = r32[brow + c2];
                bU.u[p] = u32p[brow + c2];
                bN.u[p] = n32[brow + c2];
            }
            accR = __builtin_amdgcn_wmma_f32_16x16x32_bf16(
                false, a.v, false, bR.v, (short)0, accR, false, false);
            accU = __builtin_amdgcn_wmma_f32_16x16x32_bf16(
                false, a.v, false, bU.v, (short)0, accU, false, false);
            accN = __builtin_amdgcn_wmma_f32_16x16x32_bf16(
                false, a.v, false, bN.v, (short)0, accN, false, false);
        }
        __syncthreads();
    }

    // ---- fused GRU update on the 16x16 C tile ----
    #pragma unroll
    for (int r = 0; r < 8; r++) {
        int m = b0 + r + 8 * hi;
        int n = n0 + wv * 16 + lo;
        size_t idx = (size_t)m * HSZ + n;
        float rg = sigmoidf_fast(Xr[idx] + accR[r]);
        float ug = sigmoidf_fast(Xu[idx] + accU[r]);
        float ng = tanhf(Xn[idx] + rg * accN[r]);
        float ho = h_in[idx];
        float hn = ug * ho + (1.0f - ug) * ng;
        out_s[idx] = hn;
        h_out[idx] = hn;
    }
}

// ---------------------------------------------------------------------------
extern "C" void kernel_launch(void* const* d_in, const int* in_sizes, int n_in,
                              void* d_out, int out_size, void* d_ws, size_t ws_size,
                              hipStream_t stream) {
    (void)in_sizes; (void)n_in; (void)out_size; (void)ws_size;

    const float* input  = (const float*)d_in[0]; // [SEQ][BATCH][ISZ]
    const float* hidden = (const float*)d_in[1]; // [BATCH][HSZ]
    const float* Wr = (const float*)d_in[2];
    const float* br = (const float*)d_in[3];
    const float* Wu = (const float*)d_in[4];
    const float* bu = (const float*)d_in[5];
    const float* Wn = (const float*)d_in[6];
    const float* bn = (const float*)d_in[7];

    // Workspace layout
    char* ws = (char*)d_ws;
    size_t off = 0;
    unsigned short* in_bf = (unsigned short*)(ws + off); off += (size_t)MTOT * ISZ * 2;   // 64 MB
    unsigned short* wx    = (unsigned short*)(ws + off); off += 3ULL * HSZ * ISZ * 2;     // 6 MB
    unsigned short* wh    = (unsigned short*)(ws + off); off += 3ULL * HSZ * HSZ * 2;     // 6 MB
    float* X  = (float*)(ws + off); off += 3ULL * MTOT * HSZ * 4;                         // 384 MB
    float* h0 = (float*)(ws + off); off += (size_t)BATCH * HSZ * 4;
    float* h1 = (float*)(ws + off); off += (size_t)BATCH * HSZ * 4;

    // h state init (f32)
    hipMemcpyAsync(h0, hidden, (size_t)BATCH * HSZ * 4,
                   hipMemcpyDeviceToDevice, stream);

    // One-time conversions to bf16
    k_f32_to_bf16<<<2048, 256, 0, stream>>>(input, in_bf, MTOT * ISZ);
    k_split_weights<<<2048, 256, 0, stream>>>(Wr, Wu, Wn, wx, wh);

    // Big parallel GEMM: pre-activations for all timesteps, 3 gates
    dim3 gg(HSZ / 64, MTOT / 32, 3);
    k_gemm_xgates<<<gg, 128, 0, stream>>>(in_bf, wx, br, bu, bn, X);

    // Sequential scan: 512 dependent step kernels, ping-pong h state
    float* outputs = (float*)d_out;
    float* hcur = h0;
    float* hnext = h1;
    dim3 gs(HSZ / 64, BATCH / 16);
    for (int s = 0; s < SEQ; s++) {
        const float* Xr = X + 0ULL * MTOT * HSZ + (size_t)s * BATCH * HSZ;
        const float* Xu = X + 1ULL * MTOT * HSZ + (size_t)s * BATCH * HSZ;
        const float* Xn = X + 2ULL * MTOT * HSZ + (size_t)s * BATCH * HSZ;
        float* out_s = outputs + (size_t)s * BATCH * HSZ;
        k_gru_step<<<gs, 128, 0, stream>>>(hcur, wh, Xr, Xu, Xn, out_s, hnext);
        float* t = hcur; hcur = hnext; hnext = t;
    }

    // hidden_final follows outputs in d_out
    hipMemcpyAsync(outputs + (size_t)SEQ * BATCH * HSZ, hcur,
                   (size_t)BATCH * HSZ * 4, hipMemcpyDeviceToDevice, stream);
}
